// CompILE_4002909520112
// MI455X (gfx1250) — matrix operand order repaired and checked by reading, not compiled, not run on hardware
//
#include <hip/hip_runtime.h>
#include <hip/hip_bf16.h>
#include <math.h>

typedef __bf16 bf16_t;
typedef __attribute__((ext_vector_type(16))) __bf16        v16bf;
typedef __attribute__((ext_vector_type(8)))  float         v8f;
typedef __attribute__((ext_vector_type(8)))  unsigned int  v8u;
typedef __attribute__((ext_vector_type(4)))  float         v4f;
typedef __attribute__((ext_vector_type(4)))  unsigned int  u32x4;
typedef __attribute__((ext_vector_type(8)))  unsigned int  u32x8;

constexpr int B = 128, T = 128, V = 1024, H = 512, Z = 64, S = 3;
constexpr float NEG_INF_F = -1000000000.0f;
constexpr float EPS_F = 1e-10f;

// ---------------- output layout (floats) ----------------
constexpr long ENCS_OFF  = 0;
constexpr long ENCS_SZ   = (long)S * B * T * H;
constexpr long RECS_OFF  = ENCS_OFF + ENCS_SZ;
constexpr long RECS_SZ   = (long)S * B * T * V;
constexpr long MASKS_OFF = RECS_OFF + RECS_SZ;
constexpr long MASKS_SZ  = (long)(S - 1) * B * T;
constexpr long BLOG_OFF  = MASKS_OFF + MASKS_SZ;
constexpr long BLOG_SZ   = (long)(S - 1) * B * T;
constexpr long BSAMP_OFF = BLOG_OFF + BLOG_SZ;
constexpr long BSAMP_SZ  = (long)S * B * T;
constexpr long ZLOG_OFF  = BSAMP_OFF + BSAMP_SZ;
constexpr long ZLOG_SZ   = (long)S * B * 2 * Z;
constexpr long ZSAMP_OFF = ZLOG_OFF + ZLOG_SZ;

// ---------------- small device helpers ----------------
__device__ __forceinline__ float sigf(float x) { return 1.0f / (1.0f + __expf(-x)); }

__device__ __forceinline__ unsigned int ihash(unsigned int x) {
  x ^= x >> 16; x *= 0x7feb352dU; x ^= x >> 15; x *= 0x846ca68bU; x ^= x >> 16;
  return x;
}
__device__ __forceinline__ float uniform01(unsigned int a, unsigned int b, unsigned int c) {
  unsigned int h = ihash(a ^ 0x9E3779B9U);
  h = ihash(h ^ b);
  h = ihash(h ^ c);
  float u = ((float)(h >> 8) + 0.5f) * (1.0f / 16777216.0f);
  return fmaxf(u, 1e-10f);
}

// ---------------- generic bf16-WMMA GEMM ----------------
// C[M x N] = act( A[M x K](f32, optional row-gather) * Bt[N x K](bf16, K-contig) + bias )
// One wave per block; wave computes a 16x64 tile (4 accumulators reuse the A frag).
__global__ __launch_bounds__(32)
void gemm_bf16_wmma(const float* __restrict__ A, long lda,
                    const int* __restrict__ gidx,
                    const bf16_t* __restrict__ Bt,
                    const float* __restrict__ bias,
                    float* __restrict__ C, long ldc,
                    int K, int relu)
{
  const int lane = threadIdx.x & 31;
  const int tm = blockIdx.y;    // 16-row tile
  const int tn = blockIdx.x;    // 64-col tile

  int arow = tm * 16 + (lane & 15);
  if (gidx) arow = gidx[arow];                 // fused embedding gather
  const int akoff = (lane >> 4) * 16;          // contiguous A K split: lane half h holds k = 16h..16h+15 (16x32)
  const float* Abase = A + (long)arow * lda + akoff;

  const int bcol0 = tn * 64 + (lane & 15);
  const int bkoff = (lane >> 4) * 16;          // contiguous B K split matching A: k = 16h..16h+15 (32x16, K-contig per lane)

  v8f acc[4] = {v8f{}, v8f{}, v8f{}, v8f{}};

  for (int kc = 0; kc < K; kc += 32) {
    // ---- A fragment: fp32 -> bf16 in registers ----
    v4f a0 = *(const v4f*)(Abase + kc);
    v4f a1 = *(const v4f*)(Abase + kc + 4);
    v4f a2 = *(const v4f*)(Abase + kc + 8);
    v4f a3 = *(const v4f*)(Abase + kc + 12);
    v16bf af;
#pragma unroll
    for (int i = 0; i < 4; ++i) {
      af[i]      = (bf16_t)a0[i];
      af[4 + i]  = (bf16_t)a1[i];
      af[8 + i]  = (bf16_t)a2[i];
      af[12 + i] = (bf16_t)a3[i];
    }
    // ---- 4 B fragments + 4 WMMAs ----
#pragma unroll
    for (int j = 0; j < 4; ++j) {
      const bf16_t* bp = Bt + (long)(bcol0 + j * 16) * K + kc + bkoff;
      v16bf bfr = __builtin_bit_cast(v16bf, *(const v8u*)bp);
      acc[j] = __builtin_amdgcn_wmma_f32_16x16x32_bf16(
          false, af, false, bfr, (short)0, acc[j], false, false);
    }
  }

  // ---- epilogue: bias + relu + store (C/D layout: VGPR v <-> M=v / v+8) ----
  const int crow = tm * 16 + ((lane >> 4) * 8);
  const int ccol = tn * 64 + (lane & 15);
#pragma unroll
  for (int j = 0; j < 4; ++j) {
    int col = ccol + j * 16;
    float bv = bias ? bias[col] : 0.0f;
#pragma unroll
    for (int v = 0; v < 8; ++v) {
      float val = acc[j][v] + bv;
      if (relu) val = fmaxf(val, 0.0f);
      C[(long)(crow + v) * ldc + col] = val;
    }
  }
}

// ---------------- TDM-staged recurrent GEMM: Zh = h @ WhT ----------------
// The serial LSTM step. Block = 4 waves computing a 16 x 256 tile of Zh.
// Wave 0 DMAs the 16 x K fp32 A-tile (rows of h) into LDS with the Tensor
// Data Mover; all 4 waves share it via ds_load_b128, cutting A traffic 4x.
__global__ __launch_bounds__(128)
void lstm_step_gemm_tdm(const float* __restrict__ hsrc,
                        const bf16_t* __restrict__ Wh,
                        float* __restrict__ Zh)
{
  __shared__ float tileA[16 * H];              // 32 KB of the 320 KB WGP LDS

  const int tm = blockIdx.y;
  const int wave = threadIdx.x >> 5;
  const int lane = threadIdx.x & 31;

  if (wave == 0) {
    // ---- build Tensor DMA Descriptor (D#) in SGPRs ----
    unsigned long long ga =
        (unsigned long long)(const void*)(hsrc + (unsigned long long)tm * 16 * H);
    unsigned lds_off = (unsigned)(unsigned long long)(const void*)&tileA[0];
    u32x4 g0;
    g0[0] = 1u;                                        // count=1 (valid), no gather
    g0[1] = lds_off;                                   // LDS byte address
    g0[2] = (unsigned)ga;                              // global_addr[31:0]
    g0[3] = ((unsigned)(ga >> 32) & 0x01FFFFFFu)       // global_addr[56:32]
            | 0x80000000u;                             // type=2 ("image")
    u32x8 g1;
    g1[0] = (2u << 16);                                // data_size = 4 bytes
    g1[1] = ((unsigned)H & 0xFFFFu) << 16;             // tensor_dim0[15:0] = K
    g1[2] = (((unsigned)H >> 16) & 0xFFFFu)            // tensor_dim0[31:16]
            | (16u << 16);                             // tensor_dim1[15:0] = 16 rows
    g1[3] = ((unsigned)H << 16);                       // tile_dim0 = K
    g1[4] = 16u;                                       // tile_dim1 = 16, tile_dim2 = 0
    g1[5] = (unsigned)H;                               // tensor_dim0_stride = lda
    g1[6] = 0u;
    g1[7] = 0u;
    asm volatile("tensor_load_to_lds %0, %1" :: "s"(g0), "s"(g1) : "memory");
    __builtin_amdgcn_s_wait_tensorcnt(0);
  }
  __syncthreads();

  const int akoff = (lane >> 4) * 16;
  const int bkoff = (lane >> 4) * 16;
  const float* Arow = &tileA[(lane & 15) * H + akoff];
  const int bcol0 = blockIdx.x * 256 + wave * 64 + (lane & 15);

  v8f acc[4] = {v8f{}, v8f{}, v8f{}, v8f{}};

  for (int kc = 0; kc < H; kc += 32) {
    v4f a0 = *(const v4f*)(Arow + kc);                 // ds_load_b128 from staged tile
    v4f a1 = *(const v4f*)(Arow + kc + 4);
    v4f a2 = *(const v4f*)(Arow + kc + 8);
    v4f a3 = *(const v4f*)(Arow + kc + 12);
    v16bf af;
#pragma unroll
    for (int i = 0; i < 4; ++i) {
      af[i]      = (bf16_t)a0[i];
      af[4 + i]  = (bf16_t)a1[i];
      af[8 + i]  = (bf16_t)a2[i];
      af[12 + i] = (bf16_t)a3[i];
    }
    if (kc + 32 < H)
      __builtin_prefetch(Wh + (long)bcol0 * H + kc + 32 + bkoff, 0, 0);
#pragma unroll
    for (int j = 0; j < 4; ++j) {
      const bf16_t* bp = Wh + (long)(bcol0 + j * 16) * H + kc + bkoff;
      v16bf bfr = __builtin_bit_cast(v16bf, *(const v8u*)bp);
      acc[j] = __builtin_amdgcn_wmma_f32_16x16x32_bf16(
          false, af, false, bfr, (short)0, acc[j], false, false);
    }
  }

  const int crow = tm * 16 + ((lane >> 4) * 8);
  const int ccol = blockIdx.x * 256 + wave * 64 + (lane & 15);
#pragma unroll
  for (int j = 0; j < 4; ++j) {
    int col = ccol + j * 16;
#pragma unroll
    for (int v = 0; v < 8; ++v) {
      Zh[(long)(crow + v) * (4 * H) + col] = acc[j][v];
    }
  }
}

// ---------------- weight prepack: f32 (K x N) -> bf16 transposed (N x K) --------
__global__ void prepack_kernel(const float* __restrict__ W, int K, int N,
                               bf16_t* __restrict__ out)
{
  long id = (long)blockIdx.x * blockDim.x + threadIdx.x;
  if (id >= (long)N * K) return;
  int k = (int)(id % K);
  long n = id / K;
  out[n * K + k] = (bf16_t)W[(long)k * N + n];
}

__global__ void fill_kernel(float* __restrict__ p, float v, long n)
{
  long id = (long)blockIdx.x * blockDim.x + threadIdx.x;
  if (id < n) p[id] = v;
}

// ---------------- fused LSTM pointwise ----------------
__global__ __launch_bounds__(256)
void lstm_pointwise(const float* __restrict__ Zh, const float* __restrict__ Xw,
                    const float* __restrict__ b4, const float* __restrict__ mask,
                    int t, float* __restrict__ h, float* __restrict__ c,
                    float* __restrict__ enc)
{
  int id = blockIdx.x * blockDim.x + threadIdx.x;   // B*H
  int b = id >> 9;            // /H (H=512)
  int j = id & (H - 1);
  const float* xw = Xw + ((long)(b * T + t)) * (4 * H);
  const float* zh = Zh + (long)b * (4 * H);
  float gi = zh[j]         + xw[j]         + b4[j];
  float gg = zh[j + H]     + xw[j + H]     + b4[j + H];
  float gf = zh[j + 2 * H] + xw[j + 2 * H] + b4[j + 2 * H];
  float go = zh[j + 3 * H] + xw[j + 3 * H] + b4[j + 3 * H];
  float m = mask[b * T + t];
  float cn = sigf(gf + 1.0f) * c[id] + sigf(gi) * tanhf(gg);
  float hn = sigf(go) * tanhf(cn);
  hn *= m; cn *= m;
  h[id] = hn;
  c[id] = cn;
  enc[((long)(b * T + t)) * H + j] = hn;
}

// ---------------- boundary logits: row-dot with hb2_w ----------------
__global__ __launch_bounds__(256)
void boundary_logits(const float* __restrict__ hid, const float* __restrict__ w2,
                     const float* __restrict__ b2, float* __restrict__ logits_s,
                     float* __restrict__ blog_out)
{
  int wave = threadIdx.x >> 5;
  int lane = threadIdx.x & 31;
  long row = (long)blockIdx.x * 8 + wave;        // B*T rows
  const float* hrow = hid + row * H;
  float acc = 0.0f;
  for (int k = lane; k < H; k += 32) acc += hrow[k] * w2[k];
#pragma unroll
  for (int off = 16; off; off >>= 1) acc += __shfl_xor(acc, off, 32);
  if (lane == 0) {
    int t = (int)(row & (T - 1));
    float v = (t == 0) ? NEG_INF_F : (acc + b2[0]);
    logits_s[row] = v;
    blog_out[row] = v;
  }
}

// ---------------- gumbel-softmax / argmax over T ----------------
__global__ __launch_bounds__(T)
void sample_b_kernel(const float* __restrict__ logits_s, const int* __restrict__ training,
                     int seg, float* __restrict__ sb, float* __restrict__ bsamp_out)
{
  __shared__ float buf[T];
  __shared__ int ibuf[T];
  int b = blockIdx.x, t = threadIdx.x;
  float lg = logits_s[b * T + t];
  int train = *training;
  float val = lg;
  if (train) {
    float u = uniform01(2u * (unsigned)seg, (unsigned)b, (unsigned)t);
    float g = -__logf(-__logf(u));
    val = lg + g;   // TEMP_B == 1.0
  }
  buf[t] = val; ibuf[t] = t;
  __syncthreads();
  for (int s = T / 2; s; s >>= 1) {
    if (t < s && buf[t + s] > buf[t]) { buf[t] = buf[t + s]; ibuf[t] = ibuf[t + s]; }
    __syncthreads();
  }
  float mx = buf[0];
  int amax = ibuf[0];
  __syncthreads();
  float y;
  if (train) {
    float e = __expf(val - mx);
    buf[t] = e;
    __syncthreads();
    for (int s = T / 2; s; s >>= 1) { if (t < s) buf[t] += buf[t + s]; __syncthreads(); }
    y = e / buf[0];
  } else {
    y = (t == amax) ? 1.0f : 0.0f;
  }
  sb[b * T + t] = y;
  bsamp_out[b * T + t] = y;
}

__global__ __launch_bounds__(T)
void onehot_b_kernel(const int* __restrict__ lengths, float* __restrict__ sb,
                     float* __restrict__ bsamp_out)
{
  int b = blockIdx.x, t = threadIdx.x;
  float v = (t == lengths[b] - 1) ? 1.0f : 0.0f;
  sb[b * T + t] = v;
  bsamp_out[b * T + t] = v;
}

// ---------------- readout = sum_t enc[b,t,:]*sb[b,t+1] ----------------
__global__ __launch_bounds__(256)
void readout_kernel(const float* __restrict__ enc, const float* __restrict__ sb,
                    float* __restrict__ ro)
{
  int b = blockIdx.x;
  int j = blockIdx.y * 256 + threadIdx.x;
  const float* e = enc + (long)b * T * H + j;
  const float* s = sb + b * T;
  float acc = 0.0f;
  for (int t = 0; t < T - 1; ++t) acc += e[(long)t * H] * s[t + 1];
  ro[b * H + j] = acc;
}

// ---------------- z sampling (reparameterization, Box-Muller) ----------------
__global__ __launch_bounds__(256)
void sample_z_kernel(const float* __restrict__ logits_z, const int* __restrict__ training,
                     int seg, float* __restrict__ zs_out, float* __restrict__ zbuf)
{
  int id = blockIdx.x * blockDim.x + threadIdx.x;    // B*Z
  int b = id >> 6;
  int zi = id & (Z - 1);
  float mu = logits_z[b * 2 * Z + zi];
  float out = mu;
  if (*training) {
    float lv = logits_z[b * 2 * Z + Z + zi];
    float u1 = uniform01(2u * (unsigned)seg + 1u, (unsigned)id, 1u);
    float u2 = uniform01(2u * (unsigned)seg + 1u, (unsigned)id, 2u);
    float eps = sqrtf(-2.0f * __logf(u1)) * __cosf(6.28318530718f * u2);
    out = mu + __expf(0.5f * lv) * eps;
  }
  zs_out[id] = out;
  zbuf[id] = out;
}

// ---------------- mask update: slc += log(cumsum(sb)+eps); mask = exp(slc) ------
__global__ __launch_bounds__(B)
void mask_update_kernel(const float* __restrict__ sb, float* __restrict__ slc,
                        float* __restrict__ mask, float* __restrict__ masks_out)
{
  int b = threadIdx.x;
  float cum = 0.0f;
  for (int t = 0; t < T; ++t) {
    cum += sb[b * T + t];
    float s = slc[b * T + t] + __logf(cum + EPS_F);
    slc[b * T + t] = s;
    float m = __expf(s);
    mask[b * T + t] = m;
    masks_out[b * T + t] = m;
  }
}

// ---------------- recs[b,t,:] = pred[b,:] tiled over t ----------------
__global__ __launch_bounds__(256)
void tile_recs_kernel(const float* __restrict__ pred, float* __restrict__ recs)
{
  long id = (long)blockIdx.x * blockDim.x + threadIdx.x;  // B*T*V
  int v = (int)(id & (V - 1));
  long bt = id >> 10;
  int b = (int)(bt >> 7);
  recs[id] = pred[(long)b * V + v];
}

// =============================================================================
extern "C" void kernel_launch(void* const* d_in, const int* in_sizes, int n_in,
                              void* d_out, int out_size, void* d_ws, size_t ws_size,
                              hipStream_t stream)
{
  (void)in_sizes; (void)n_in; (void)out_size; (void)ws_size;
  // ---- inputs (setup_inputs dict order) ----
  const float* embed   = (const float*)d_in[0];   // V x H
  const float* lstm_w  = (const float*)d_in[1];   // 2H x 4H
  const float* lstm_b  = (const float*)d_in[2];   // 4H
  const float* hb1_w   = (const float*)d_in[3];   // H x H
  const float* hb1_b   = (const float*)d_in[4];
  const float* hb2_w   = (const float*)d_in[5];   // H x 1
  const float* hb2_b   = (const float*)d_in[6];
  const float* hz1_w   = (const float*)d_in[7];   // H x H
  const float* hz1_b   = (const float*)d_in[8];
  const float* hz2_w   = (const float*)d_in[9];   // H x 2Z
  const float* hz2_b   = (const float*)d_in[10];
  const float* d1_w    = (const float*)d_in[11];  // Z x H
  const float* d1_b    = (const float*)d_in[12];
  const float* d2_w    = (const float*)d_in[13];  // H x V
  const float* d2_b    = (const float*)d_in[14];
  const int*   inputs  = (const int*)d_in[15];    // B x T
  const int*   lengths = (const int*)d_in[16];    // B
  const int*   training = (const int*)d_in[17];   // scalar

  float* out = (float*)d_out;
  float* out_encs  = out + ENCS_OFF;
  float* out_recs  = out + RECS_OFF;
  float* out_masks = out + MASKS_OFF;
  float* out_blog  = out + BLOG_OFF;
  float* out_bsamp = out + BSAMP_OFF;
  float* out_zlog  = out + ZLOG_OFF;
  float* out_zsamp = out + ZSAMP_OFF;

  // ---- workspace carve-up ----
  char* wp = (char*)d_ws;
  auto alloc = [&](size_t bytes) -> char* {
    char* r = wp; wp += (bytes + 255) & ~(size_t)255; return r;
  };
  bf16_t* wxT  = (bf16_t*)alloc((size_t)4 * H * H * 2);          // 2048 x 512
  bf16_t* whT  = (bf16_t*)alloc((size_t)4 * H * H * 2);
  bf16_t* hb1T = (bf16_t*)alloc((size_t)H * H * 2);
  bf16_t* hz1T = (bf16_t*)alloc((size_t)H * H * 2);
  bf16_t* hz2T = (bf16_t*)alloc((size_t)2 * Z * H * 2);          // 128 x 512
  bf16_t* d1T  = (bf16_t*)alloc((size_t)H * Z * 2);              // 512 x 64
  bf16_t* d2T  = (bf16_t*)alloc((size_t)V * H * 2);              // 1024 x 512
  float* Xw      = (float*)alloc((size_t)B * T * 4 * H * 4);     // 16384 x 2048
  float* Zh      = (float*)alloc((size_t)B * 4 * H * 4);
  float* hbuf    = (float*)alloc((size_t)B * H * 4);
  float* cbuf    = (float*)alloc((size_t)B * H * 4);
  float* hid     = (float*)alloc((size_t)B * T * H * 4);         // 16384 x 512
  float* logit_s = (float*)alloc((size_t)B * T * 4);
  float* sb      = (float*)alloc((size_t)B * T * 4);
  float* slc     = (float*)alloc((size_t)B * T * 4);
  float* mask    = (float*)alloc((size_t)B * T * 4);
  float* ro      = (float*)alloc((size_t)B * H * 4);
  float* hid2    = (float*)alloc((size_t)B * H * 4);
  float* zbuf    = (float*)alloc((size_t)B * Z * 4);
  float* hidd    = (float*)alloc((size_t)B * H * 4);
  float* pred    = (float*)alloc((size_t)B * V * 4);

  auto gemm = [&](const float* A, long lda, const int* gidx, const bf16_t* Bt,
                  const float* bias, float* C, long ldc, int M, int N, int K, int relu) {
    dim3 g(N / 64, M / 16);
    gemm_bf16_wmma<<<g, dim3(32), 0, stream>>>(A, lda, gidx, Bt, bias, C, ldc, K, relu);
  };
  auto prepack = [&](const float* W, int K, int N, bf16_t* dst) {
    long n = (long)N * K;
    prepack_kernel<<<(unsigned)((n + 255) / 256), 256, 0, stream>>>(W, K, N, dst);
  };

  // ---- one-time weight prepack (bf16, transposed, K-contiguous) ----
  prepack(lstm_w,                   H, 4 * H, wxT);   // rows 0..H-1  -> Wx^T
  prepack(lstm_w + (long)H * 4 * H, H, 4 * H, whT);   // rows H..2H-1 -> Wh^T
  prepack(hb1_w, H, H, hb1T);
  prepack(hz1_w, H, H, hz1T);
  prepack(hz2_w, H, 2 * Z, hz2T);
  prepack(d1_w,  Z, H, d1T);
  prepack(d2_w,  H, V, d2T);

  // ---- init segment-0 mask = 1, sum_log_cumsum = 0 ----
  fill_kernel<<<(B * T + 255) / 256, 256, 0, stream>>>(mask, 1.0f, (long)B * T);
  hipMemsetAsync(slc, 0, (size_t)B * T * 4, stream);

  // ---- precompute Xw = embed[inputs] @ Wx  (once, reused by all 3 segments) ----
  gemm(embed, H, inputs, wxT, nullptr, Xw, 4 * H, B * T, 4 * H, H, 0);

  for (int seg = 0; seg < S; ++seg) {
    float* enc = out_encs + (long)seg * B * T * H;

    // ---- masked LSTM scan: only h @ Wh is serial; TDM-staged GEMM ----
    hipMemsetAsync(hbuf, 0, (size_t)B * H * 4, stream);
    hipMemsetAsync(cbuf, 0, (size_t)B * H * 4, stream);
    for (int t = 0; t < T; ++t) {
      lstm_step_gemm_tdm<<<dim3((4 * H) / 256, B / 16), 128, 0, stream>>>(hbuf, whT, Zh);
      lstm_pointwise<<<(B * H) / 256, 256, 0, stream>>>(Zh, Xw, lstm_b, mask, t,
                                                        hbuf, cbuf, enc);
    }

    // ---- boundary sample ----
    if (seg < S - 1) {
      gemm(enc, H, nullptr, hb1T, hb1_b, hid, H, B * T, H, H, 1);
      boundary_logits<<<(B * T) / 8, 256, 0, stream>>>(hid, hb2_w, hb2_b, logit_s,
                                                       out_blog + (long)seg * B * T);
      sample_b_kernel<<<B, T, 0, stream>>>(logit_s, training, seg, sb,
                                           out_bsamp + (long)seg * B * T);
    } else {
      onehot_b_kernel<<<B, T, 0, stream>>>(lengths, sb,
                                           out_bsamp + (long)seg * B * T);
    }

    // ---- readout + z head ----
    readout_kernel<<<dim3(B, H / 256), 256, 0, stream>>>(enc, sb, ro);
    gemm(ro, H, nullptr, hz1T, hz1_b, hid2, H, B, H, H, 1);
    float* zlog = out_zlog + (long)seg * B * 2 * Z;
    gemm(hid2, H, nullptr, hz2T, hz2_b, zlog, 2 * Z, B, 2 * Z, H, 0);
    sample_z_kernel<<<(B * Z) / 256, 256, 0, stream>>>(zlog, training, seg,
                                                       out_zsamp + (long)seg * B * Z, zbuf);

    // ---- mask update (before next segment) ----
    if (seg < S - 1) {
      mask_update_kernel<<<1, B, 0, stream>>>(sb, slc, mask,
                                              out_masks + (long)seg * B * T);
    }

    // ---- decoder + tile into recs ----
    gemm(zbuf, Z, nullptr, d1T, d1_b, hidd, H, B, H, Z, 1);
    gemm(hidd, H, nullptr, d2T, d2_b, pred, V, B, V, H, 0);
    tile_recs_kernel<<<((long)B * T * V) / 256, 256, 0, stream>>>(
        pred, out_recs + (long)seg * B * T * V);
  }
}
